// DiscriminatorCNN_27788438405339
// MI455X (gfx1250) — matrix operand order repaired and checked by reading, not compiled, not run on hardware
//
#include <hip/hip_runtime.h>
#include <hip/hip_bf16.h>

#define B_TOT   65536
#define S_DIM   20000
#define D_DIM   300
#define PFD     12
#define LFD     8
#define A_DIM   8
#define C_IN    21      // PF + LF + 1
#define K1PAD   64      // fc1 K: 38 -> 64
#define K2PAD   128     // fc2 K: 120 -> 128
#define N1PAD   128     // fc1 N: 120 -> 128 (8 tiles)
#define N2PAD   96      // fc2 N: 84 -> 96 (6 tiles)

typedef __attribute__((ext_vector_type(16))) _Float16 v16h;
typedef __attribute__((ext_vector_type(8)))  _Float16 v8h;
typedef __attribute__((ext_vector_type(8)))  float    v8f;

__device__ __forceinline__ float lrelu(float v) { return fmaxf(v, 0.2f * v); }

// ---------------------------------------------------------------------------
// Kernel A: pack fc1/fc2 weights into WMMA B-fragment layout (f16).
// B-matrix 32x16 (16-bit), contiguous K order used here: lane l holds column n = l%16; its 16 halves hold
// K = (l/16)*16 + j for j=0..15 within the 32-wide K chunk.
// fc1: 8 N-tiles x 2 K-chunks; fc2: 6 N-tiles x 4 K-chunks.
// ---------------------------------------------------------------------------
__global__ __launch_bounds__(256) void pack_weights_kernel(
    const float* __restrict__ f1w,   // (120, 38)
    const float* __restrict__ f2w,   // (84, 120)
    _Float16* __restrict__ w1p,      // 8*2*32*16 = 8192 halves
    _Float16* __restrict__ w2p)      // 6*4*32*16 = 12288 halves
{
    int idx = blockIdx.x * 256 + threadIdx.x;
    if (idx < 8192) {
        int j  = idx & 15;
        int l  = (idx >> 4) & 31;
        int tc = idx >> 9;            // t*2 + c
        int c  = tc & 1, t = tc >> 1;
        int n  = t * 16 + (l & 15);
        int k  = c * 32 + (l >> 4) * 16 + j;
        float v = (n < 120 && k < 38) ? f1w[n * 38 + k] : 0.0f;
        w1p[idx] = (_Float16)v;
    }
    int idx2 = idx - 8192;
    if (idx2 >= 0 && idx2 < 12288) {
        int j  = idx2 & 15;
        int l  = (idx2 >> 4) & 31;
        int tc = idx2 >> 9;           // t*4 + c
        int c  = tc & 3, t = tc >> 2;
        int n  = t * 16 + (l & 15);
        int k  = c * 32 + (l >> 4) * 16 + j;
        float v = (n < 84 && k < 120) ? f2w[n * 120 + k] : 0.0f;
        w2p[idx2] = (_Float16)v;
    }
}

// ---------------------------------------------------------------------------
// Kernel B: gather + conv1 + lrelu + maxpool + conv2 + lrelu + one-hot concat.
// One thread per sample. Features staged per-thread in LDS (f16), conv
// weights cached in LDS (f32). Emits 64-half padded rows of X (f16).
// ---------------------------------------------------------------------------
__global__ __launch_bounds__(64) void feat_conv_kernel(
    const int*   __restrict__ state,
    const int*   __restrict__ des,
    const int*   __restrict__ act,
    const int*   __restrict__ asp,    // (S, 9)
    const int*   __restrict__ pmp,    // (S, 9)
    const float* __restrict__ pfeat,  // (S, 300, 12)
    const float* __restrict__ lfeat,  // (S, 8)
    const float* __restrict__ w1,     // (20, 21, 3, 3)
    const float* __restrict__ b1,     // (20,)
    const float* __restrict__ w2,     // (30, 20, 2, 2)
    const float* __restrict__ b2,     // (30,)
    _Float16* __restrict__ X)         // (B, 64)
{
    __shared__ float    sW1[20 * 21 * 9];
    __shared__ float    sB1[20];
    __shared__ float    sW2[30 * 20 * 4];
    __shared__ float    sB2[30];
    __shared__ _Float16 sFeat[64 * 192];   // 21*9 = 189 per thread, pad 192

    const int tid = threadIdx.x;
    for (int i = tid; i < 20 * 21 * 9; i += 64) sW1[i] = w1[i];
    for (int i = tid; i < 30 * 20 * 4; i += 64) sW2[i] = w2[i];
    if (tid < 20) sB1[tid] = b1[tid];
    if (tid < 30) sB2[tid] = b2[tid];

    const int gid = blockIdx.x * 64 + tid;
    const int s = state[gid], d = des[gid], a = act[gid];

    int neigh[9], mask[9];
    #pragma unroll
    for (int j = 0; j < 9; ++j) { neigh[j] = asp[s * 9 + j]; mask[j] = pmp[s * 9 + j]; }

    // feat layout in LDS: channel-major, spatial p = h*3+w, NEW_INDEX applied.
    const int NEW_IDX[9] = {7, 0, 1, 6, 8, 2, 5, 4, 3};
    _Float16* f = &sFeat[tid * 192];
    #pragma unroll
    for (int p = 0; p < 9; ++p) {
        const int j = NEW_IDX[p];
        const int n = neigh[j];
        const float* prow = pfeat + ((long)n * D_DIM + d) * PFD;
        #pragma unroll
        for (int c = 0; c < PFD; ++c) f[c * 9 + p] = (_Float16)prow[c];
        const float* lrow = lfeat + (long)n * LFD;
        #pragma unroll
        for (int c = 0; c < LFD; ++c) f[(PFD + c) * 9 + p] = (_Float16)lrow[c];
        f[(PFD + LFD) * 9 + p] = (_Float16)(float)mask[j];
    }
    __syncthreads();

    float out2[30];
    #pragma unroll
    for (int k = 0; k < 30; ++k) out2[k] = sB2[k];

    #pragma unroll 1
    for (int o = 0; o < 20; ++o) {
        float plane[9];
        #pragma unroll
        for (int q = 0; q < 9; ++q) plane[q] = sB1[o];
        const float* wo = &sW1[o * 21 * 9];
        #pragma unroll 1
        for (int c = 0; c < C_IN; ++c) {
            float xc[9];
            #pragma unroll
            for (int q = 0; q < 9; ++q) xc[q] = (float)f[c * 9 + q];
            const float* wc = &wo[c * 9];
            #pragma unroll
            for (int h = 0; h < 3; ++h) {
                #pragma unroll
                for (int w = 0; w < 3; ++w) {
                    float acc = 0.0f;
                    #pragma unroll
                    for (int kh = 0; kh < 3; ++kh) {
                        const int hh = h + kh - 1;
                        if (hh < 0 || hh > 2) continue;
                        #pragma unroll
                        for (int kw = 0; kw < 3; ++kw) {
                            const int ww = w + kw - 1;
                            if (ww < 0 || ww > 2) continue;
                            acc += xc[hh * 3 + ww] * wc[kh * 3 + kw];
                        }
                    }
                    plane[h * 3 + w] += acc;
                }
            }
        }
        #pragma unroll
        for (int q = 0; q < 9; ++q) plane[q] = lrelu(plane[q]);
        const float p00 = fmaxf(fmaxf(plane[0], plane[1]), fmaxf(plane[3], plane[4]));
        const float p01 = fmaxf(fmaxf(plane[1], plane[2]), fmaxf(plane[4], plane[5]));
        const float p10 = fmaxf(fmaxf(plane[3], plane[4]), fmaxf(plane[6], plane[7]));
        const float p11 = fmaxf(fmaxf(plane[4], plane[5]), fmaxf(plane[7], plane[8]));
        #pragma unroll 1
        for (int k = 0; k < 30; ++k) {
            const float* wk = &sW2[(k * 20 + o) * 4];
            out2[k] += p00 * wk[0] + p01 * wk[1] + p10 * wk[2] + p11 * wk[3];
        }
    }

    _Float16* xr = X + (long)gid * K1PAD;
    #pragma unroll
    for (int k = 0; k < 30; ++k) xr[k] = (_Float16)lrelu(out2[k]);
    #pragma unroll
    for (int c = 0; c < A_DIM; ++c) xr[30 + c] = (_Float16)((c == a) ? 1.0f : 0.0f);
    #pragma unroll
    for (int c = 38; c < K1PAD; ++c) xr[c] = (_Float16)0.0f;
}

// ---------------------------------------------------------------------------
// Kernel C: fc1/fc2 via v_wmma_f32_16x16x32_f16, fc3 + sigmoid in VALU.
// 8 waves per block; each wave owns a 16-sample M-tile.
// A-fragment (16-bit A 16x32, contiguous K, not ISA table order): lane l -> row m=l%16, hi=l/16;
// halves 0..15 = K(hi*16+0..15)  -> two 16B loads.
// D (f32 16x16): lane l -> col n=l%16; VGPR r -> row m = r + 8*hi.
// ---------------------------------------------------------------------------
__global__ __launch_bounds__(256) void fc_wmma_kernel(
    const _Float16* __restrict__ X,     // (B, 64)
    const _Float16* __restrict__ w1p,
    const _Float16* __restrict__ w2p,
    const float* __restrict__ f1b,      // (120,)
    const float* __restrict__ f2b,      // (84,)
    const float* __restrict__ f3w,      // (84,)
    const float* __restrict__ f3b,      // (1,)
    float* __restrict__ out)            // (B,)
{
    __shared__ _Float16 sY1[8 * 16 * K2PAD];  // 32 KB: fc1 output, f16
    __shared__ _Float16 sY2[8 * 16 * N2PAD];  // 24 KB: fc2 output, f16

    const int lane = threadIdx.x & 31;
    const int wave = threadIdx.x >> 5;
    const int ar   = lane & 15;   // A row / D column, depending on use
    const int hi   = lane >> 4;
    const int s0   = blockIdx.x * 128 + wave * 16;

    union Frag { v16h v; v8h h[2]; };

    // ---- fc1 A fragments: K = 64 -> 2 chunks of 32 ----
    v16h a1[2];
    {
        const _Float16* xrow = X + (long)(s0 + ar) * K1PAD;
        #pragma unroll
        for (int c = 0; c < 2; ++c) {
            Frag u;
            u.h[0] = *(const v8h*)(xrow + c * 32 + hi * 16);
            u.h[1] = *(const v8h*)(xrow + c * 32 + hi * 16 + 8);
            a1[c] = u.v;
        }
    }

    // ---- fc1: 8 N-tiles x 2 K-chunks of WMMA ----
    _Float16* y1 = &sY1[wave * 16 * K2PAD];
    #pragma unroll
    for (int t = 0; t < 8; ++t) {
        v8f acc = {};
        #pragma unroll
        for (int c = 0; c < 2; ++c) {
            v16h b = *(const v16h*)(w1p + ((t * 2 + c) * 32 + lane) * 16);
            acc = __builtin_amdgcn_wmma_f32_16x16x32_f16(
                false, a1[c], false, b, (short)0, acc, false, false);
        }
        const int n = t * 16 + ar;
        const float bias = (n < 120) ? f1b[n] : 0.0f;
        #pragma unroll
        for (int r = 0; r < 8; ++r)
            y1[(r + 8 * hi) * K2PAD + n] = (_Float16)lrelu(acc[r] + bias);
    }
    __syncthreads();

    // ---- fc2 A fragments from LDS: K = 128 -> 4 chunks of 32 ----
    v16h a2[4];
    {
        const _Float16* yrow = y1 + ar * K2PAD;
        #pragma unroll
        for (int c = 0; c < 4; ++c) {
            Frag u;
            u.h[0] = *(const v8h*)(yrow + c * 32 + hi * 16);
            u.h[1] = *(const v8h*)(yrow + c * 32 + hi * 16 + 8);
            a2[c] = u.v;
        }
    }

    // ---- fc2: 6 N-tiles x 4 K-chunks of WMMA ----
    _Float16* y2 = &sY2[wave * 16 * N2PAD];
    #pragma unroll
    for (int t = 0; t < 6; ++t) {
        v8f acc = {};
        #pragma unroll
        for (int c = 0; c < 4; ++c) {
            v16h b = *(const v16h*)(w2p + ((t * 4 + c) * 32 + lane) * 16);
            acc = __builtin_amdgcn_wmma_f32_16x16x32_f16(
                false, a2[c], false, b, (short)0, acc, false, false);
        }
        const int n = t * 16 + ar;
        const float bias = (n < 84) ? f2b[n] : 0.0f;
        #pragma unroll
        for (int r = 0; r < 8; ++r)
            y2[(r + 8 * hi) * N2PAD + n] = (_Float16)lrelu(acc[r] + bias);
    }
    __syncthreads();

    // ---- fc3 + sigmoid: lanes 0..15 each finish one sample ----
    if (lane < 16) {
        const _Float16* yr = y2 + lane * N2PAD;
        float acc = f3b[0];
        #pragma unroll 1
        for (int k = 0; k < 84; ++k) acc += (float)yr[k] * f3w[k];
        out[s0 + lane] = 1.0f / (1.0f + __expf(-acc));
    }
}

// ---------------------------------------------------------------------------
extern "C" void kernel_launch(void* const* d_in, const int* in_sizes, int n_in,
                              void* d_out, int out_size, void* d_ws, size_t ws_size,
                              hipStream_t stream)
{
    const int*   state = (const int*)d_in[0];
    const int*   des   = (const int*)d_in[1];
    const int*   act   = (const int*)d_in[2];
    const int*   asp   = (const int*)d_in[3];
    const int*   pmp   = (const int*)d_in[4];
    const float* pfeat = (const float*)d_in[5];
    const float* lfeat = (const float*)d_in[6];
    const float* c1w   = (const float*)d_in[7];
    const float* c1b   = (const float*)d_in[8];
    const float* c2w   = (const float*)d_in[9];
    const float* c2b   = (const float*)d_in[10];
    const float* f1w   = (const float*)d_in[11];
    const float* f1b   = (const float*)d_in[12];
    const float* f2w   = (const float*)d_in[13];
    const float* f2b   = (const float*)d_in[14];
    const float* f3w   = (const float*)d_in[15];
    const float* f3b   = (const float*)d_in[16];
    float* out = (float*)d_out;

    // Workspace layout: X (B x 64 f16) | packed fc1 B-frags | packed fc2 B-frags
    _Float16* Xh  = (_Float16*)d_ws;
    _Float16* w1p = Xh + (size_t)B_TOT * K1PAD;
    _Float16* w2p = w1p + 8192;

    pack_weights_kernel<<<(8192 + 12288 + 255) / 256, 256, 0, stream>>>(f1w, f2w, w1p, w2p);
    feat_conv_kernel<<<B_TOT / 64, 64, 0, stream>>>(state, des, act, asp, pmp,
                                                    pfeat, lfeat, c1w, c1b, c2w, c2b, Xh);
    fc_wmma_kernel<<<B_TOT / 128, 256, 0, stream>>>(Xh, w1p, w2p, f1b, f2b, f3w, f3b, out);
}
